// MultiHeadAttention_73632919323156
// MI455X (gfx1250) — matrix operand order repaired and checked by reading, not compiled, not run on hardware
//
#include <hip/hip_runtime.h>

typedef __attribute__((ext_vector_type(16))) _Float16 v16h;
typedef __attribute__((ext_vector_type(8)))  _Float16 v8h;
typedef __attribute__((ext_vector_type(8)))  float    v8f;

#define D_H   768
#define SEQ   4096
#define NHEAD 12
#define DA    64
#define BZ    2

static __device__ __forceinline__ v8f wmma_f16(v16h a, v16h b, v8f c) {
  // D = A(16x32 f16) * B(32x16 f16) + C(16x16 f32)
  return __builtin_amdgcn_wmma_f32_16x16x32_f16(false, a, false, b,
                                                (short)0, c, false, false);
}

// CDNA5 async global->LDS copy, 16 bytes per lane (tracked by ASYNCcnt).
static __device__ __forceinline__ void async_copy16(void* lds_dst, const void* gsrc) {
  unsigned int       l = (unsigned int)(uintptr_t)lds_dst;
  unsigned long long g = (unsigned long long)(uintptr_t)gsrc;
  asm volatile("global_load_async_to_lds_b128 %0, %1, off"
               :: "v"(l), "v"(g) : "memory");
}
static __device__ __forceinline__ void async_wait0() {
  asm volatile("s_wait_asynccnt 0x0" ::: "memory");
}

// ---- DPP cross-lane helpers (VALU cost, no LDS-routed ds_bpermute) --------
// lane ^= K (K in {1,2,4}) inside each 8-lane group via DPP8
template <int K>
static __device__ __forceinline__ float dpp8_xor(float x) {
  constexpr unsigned sel =
      (unsigned)(0 ^ K)        | ((unsigned)(1 ^ K) << 3)  |
      ((unsigned)(2 ^ K) << 6) | ((unsigned)(3 ^ K) << 9)  |
      ((unsigned)(4 ^ K) << 12)| ((unsigned)(5 ^ K) << 15) |
      ((unsigned)(6 ^ K) << 18)| ((unsigned)(7 ^ K) << 21);
  return __int_as_float(__builtin_amdgcn_mov_dpp8(__float_as_int(x), sel));
}
// lane ^= 8 inside each 16-lane row: row_ror:8 (DPP16 ctrl 0x128)
static __device__ __forceinline__ float dpp16_xor8(float x) {
  return __int_as_float(__builtin_amdgcn_update_dpp(
      0, __float_as_int(x), 0x128, 0xF, 0xF, true));
}
// full 16-lane-group reductions (both halves independent, as required)
static __device__ __forceinline__ float redmax16(float x) {
  x = fmaxf(x, dpp8_xor<1>(x));
  x = fmaxf(x, dpp8_xor<2>(x));
  x = fmaxf(x, dpp8_xor<4>(x));
  x = fmaxf(x, dpp16_xor8(x));
  return x;
}
static __device__ __forceinline__ float redsum16(float x) {
  x += dpp8_xor<1>(x);
  x += dpp8_xor<2>(x);
  x += dpp8_xor<4>(x);
  x += dpp16_xor8(x);
  return x;
}

// ---------------------------------------------------------------------------
// Projection GEMM: Y = X(8192x768 f32) @ W(768x768 f32), output f16 into
// head-split layout, compile-time selected:
//   DMAJOR=1 -> dst[b][h][d][s]  (Q and V; epilogue = b128 vector stores)
//   DMAJOR=0 -> dst[b][h][s][d]  (K; rows stay d-contiguous for async staging)
// ---------------------------------------------------------------------------
template <int DMAJOR>
__global__ __launch_bounds__(256) void proj_kernel(
    const float* __restrict__ X, const float* __restrict__ W,
    _Float16* __restrict__ dst, float scale)
{
  __shared__ __align__(16) _Float16 ldsW[64][34];  // [col][k]

  const int t    = threadIdx.x;
  const int lane = t & 31;
  const int wv   = t >> 5;
  const int ln16 = lane & 15;
  const int half = lane >> 4;
  const int m0 = blockIdx.x * 128 + wv * 16;
  const int n0 = blockIdx.y * 64;

  v8f acc[4] = {};
  const float* xrow = X + (size_t)(m0 + ln16) * D_H;

  for (int k0 = 0; k0 < D_H; k0 += 32) {
    __syncthreads();  // protect ldsW reads of previous slice
    {   // cooperative stage of W[k0..k0+31][n0..n0+63] -> ldsW[col][k] (f16)
      const int kk   = t >> 3;        // 0..31
      const int col0 = (t & 7) * 8;   // 0..56
      const float* wp = W + (size_t)(k0 + kk) * D_H + n0 + col0;
#pragma unroll
      for (int i = 0; i < 8; ++i) ldsW[col0 + i][kk] = (_Float16)wp[i];
    }
    __syncthreads();

    // A fragment (ISA 16-bit A 16x32 layout)
    v16h a;
    {
      const float* p = xrow + k0 + 8 * half;
#pragma unroll
      for (int i = 0; i < 8; ++i) {
        a[i]     = (_Float16)p[i];
        a[i + 8] = (_Float16)p[i + 16];
      }
    }
#pragma unroll
    for (int j = 0; j < 4; ++j) {
      const _Float16* bp = &ldsW[j * 16 + ln16][8 * half];
      v16h b;
#pragma unroll
      for (int i = 0; i < 16; ++i) b[i] = bp[i < 8 ? i : i + 8];
      acc[j] = wmma_f16(a, b, acc[j]);
    }
  }

  // C layout: element (vgpr r, lane) -> M = 8*half + r, N = ln16
#pragma unroll
  for (int j = 0; j < 4; ++j) {
    const int cg = n0 + j * 16 + ln16;  // global col (h*64 + d)
    const int hh = cg >> 6;
    const int dd = cg & (DA - 1);
    if (DMAJOR) {
      const int rg0 = m0 + 8 * half;
      const int bb  = rg0 >> 12;
      const int ss0 = rg0 & (SEQ - 1);
      v8h pk;
#pragma unroll
      for (int r = 0; r < 8; ++r) pk[r] = (_Float16)(acc[j][r] * scale);
      *(v8h*)&dst[((size_t)(bb * NHEAD + hh) * DA + dd) * SEQ + ss0] = pk;
    } else {
#pragma unroll
      for (int r = 0; r < 8; ++r) {
        const int rg = m0 + 8 * half + r;
        const int bb = rg >> 12;
        const int ss = rg & (SEQ - 1);
        dst[((size_t)(bb * NHEAD + hh) * SEQ + ss) * DA + dd] =
            (_Float16)(acc[j][r] * scale);
      }
    }
  }
}

// ---------------------------------------------------------------------------
// Flash attention: block = 8 waves, all on one (b,h), 128 queries (16/wave).
// K/V 32-key blocks double-buffered in LDS via async global->LDS b128 copies.
// Qt: [b][h][d][s] f16, pre-scaled by 1/8. Kh: [b][h][s][64]. VT: [b][h][d][s].
// ---------------------------------------------------------------------------
__global__ __launch_bounds__(256) void flash_kernel(
    const _Float16* __restrict__ Qt, const _Float16* __restrict__ Kh,
    const _Float16* __restrict__ VT, const int* __restrict__ mask,
    _Float16* __restrict__ attn)
{
  __shared__ __align__(16) _Float16 ldsK[2][32][72];  // 144B rows (36 banks)
  __shared__ __align__(16) _Float16 ldsV[2][64][40];  // 80B rows (20 banks)
  __shared__ __align__(16) _Float16 ldsP[8][16][40];  // per-wave P tile

  const int t    = threadIdx.x;
  const int lane = t & 31;
  const int wv   = t >> 5;
  const int ln16 = lane & 15;
  const int half = lane >> 4;

  const int bh = blockIdx.y;             // 0..23
  const int b  = bh / NHEAD;
  const int hh = bh % NHEAD;
  const int q0 = blockIdx.x * 128 + wv * 16;

  const _Float16* Q = Qt + (size_t)bh * DA * SEQ;
  const _Float16* K = Kh + (size_t)bh * SEQ * DA;
  const _Float16* V = VT + (size_t)bh * DA * SEQ;
  const int* mrow = mask + (size_t)b * SEQ;

  // staging roles: K block = 32 rows x 8 x 16B; V block = 64 rows x 4 x 16B
  const int krow = t >> 3, kchunk = t & 7;
  const int vrow = t >> 2, vchunk = t & 3;
  auto stage = [&](int buf, int kb) {
    async_copy16(&ldsK[buf][krow][kchunk * 8],
                 K + (size_t)(kb + krow) * DA + kchunk * 8);
    async_copy16(&ldsV[buf][vrow][vchunk * 8],
                 V + (size_t)vrow * SEQ + kb + vchunk * 8);
  };

  // Q A-fragments (loaded once; lane-coalesced gathers from d-major layout)
  v16h aq[2];
#pragma unroll
  for (int ak = 0; ak < 2; ++ak) {
#pragma unroll
    for (int i = 0; i < 8; ++i) {
      aq[ak][i]     = Q[(size_t)(ak * 32 + 8 * half + i)      * SEQ + q0 + ln16];
      aq[ak][i + 8] = Q[(size_t)(ak * 32 + 8 * half + 16 + i) * SEQ + q0 + ln16];
    }
  }

  v8f o[4] = {};
  float mrun[8], lrun[8];
#pragma unroll
  for (int r = 0; r < 8; ++r) { mrun[r] = -INFINITY; lrun[r] = 0.0f; }

  stage(0, 0);
  async_wait0();
  __syncthreads();

  for (int kb = 0; kb < SEQ; kb += 32) {
    const int cur = (kb >> 5) & 1;
    if (kb + 32 < SEQ) stage(cur ^ 1, kb + 32);   // prefetch next block

    // ---- scores S(16x32): two 16x16 tiles, K-dim = d = 64 ----
    v8f s[2] = {};
#pragma unroll
    for (int jj = 0; jj < 2; ++jj) {
#pragma unroll
      for (int ak = 0; ak < 2; ++ak) {
        const _Float16* p = &ldsK[cur][jj * 16 + ln16][ak * 32 + 8 * half];
        v16h bk;
#pragma unroll
        for (int i = 0; i < 16; ++i) bk[i] = p[i < 8 ? i : i + 8];
        s[jj] = wmma_f16(aq[ak], bk, s[jj]);
      }
    }

    // additive mask: -1e9 where mask==1 (per key, broadcast over queries)
    float madd0 = -1.0e9f * (float)mrow[kb + ln16];
    float madd1 = -1.0e9f * (float)mrow[kb + 16 + ln16];
#pragma unroll
    for (int r = 0; r < 8; ++r) { s[0][r] += madd0; s[1][r] += madd1; }

    // ---- online softmax (rows live across 16-lane halves; DPP trees) ----
    float scalef[8];
#pragma unroll
    for (int r = 0; r < 8; ++r) {
      const float bm   = redmax16(fmaxf(s[0][r], s[1][r]));
      const float mnew = fmaxf(mrun[r], bm);
      const float sc = __expf(mrun[r] - mnew);
      const float p0 = __expf(s[0][r] - mnew);
      const float p1 = __expf(s[1][r] - mnew);
      const float rs = redsum16(p0 + p1);
      lrun[r] = lrun[r] * sc + rs;
      mrun[r] = mnew;
      scalef[r] = sc;
      s[0][r] = p0;
      s[1][r] = p1;
    }

    // rescale running output accumulators
#pragma unroll
    for (int j = 0; j < 4; ++j)
#pragma unroll
      for (int r = 0; r < 8; ++r) o[j][r] *= scalef[r];

    // ---- P: C-layout -> A-layout via per-wave LDS tile ----
#pragma unroll
    for (int r = 0; r < 8; ++r) {
      ldsP[wv][8 * half + r][ln16]      = (_Float16)s[0][r];
      ldsP[wv][8 * half + r][16 + ln16] = (_Float16)s[1][r];
    }
    v16h pa;
    {
      const _Float16* pr = &ldsP[wv][ln16][8 * half];
#pragma unroll
      for (int i = 0; i < 8; ++i) { pa[i] = pr[i]; pa[i + 8] = pr[16 + i]; }
    }

    // ---- O(16x64) += P(16x32) . V(32x64) ----
#pragma unroll
    for (int jd = 0; jd < 4; ++jd) {
      const _Float16* vp = &ldsV[cur][jd * 16 + ln16][8 * half];
      v16h bv;
#pragma unroll
      for (int i = 0; i < 16; ++i) bv[i] = vp[i < 8 ? i : i + 8];
      o[jd] = wmma_f16(pa, bv, o[jd]);
    }

    async_wait0();     // next block's staging complete
    __syncthreads();   // everyone done reading cur before it is refilled
  }

  // ---- finalize: divide by softmax denominator, write f16 ----
#pragma unroll
  for (int r = 0; r < 8; ++r) {
    const float inv = 1.0f / lrun[r];
    const int qg = q0 + 8 * half + r;
    _Float16* orow = attn + ((size_t)b * SEQ + qg) * D_H + hh * DA;
#pragma unroll
    for (int jd = 0; jd < 4; ++jd)
      orow[jd * 16 + ln16] = (_Float16)(o[jd][r] * inv);
  }
}

// ---------------------------------------------------------------------------
// Output projection: out = attn(8192x768 f16) @ w_o(768x768 f32) + b_o, f32.
// ---------------------------------------------------------------------------
__global__ __launch_bounds__(256) void oproj_kernel(
    const _Float16* __restrict__ A, const float* __restrict__ W,
    const float* __restrict__ bias, float* __restrict__ out)
{
  __shared__ __align__(16) _Float16 ldsW[64][34];

  const int t    = threadIdx.x;
  const int lane = t & 31;
  const int wv   = t >> 5;
  const int ln16 = lane & 15;
  const int half = lane >> 4;
  const int m0 = blockIdx.x * 128 + wv * 16;
  const int n0 = blockIdx.y * 64;

  v8f acc[4] = {};
  const _Float16* arow = A + (size_t)(m0 + ln16) * D_H;

  for (int k0 = 0; k0 < D_H; k0 += 32) {
    __syncthreads();
    {
      const int kk   = t >> 3;
      const int col0 = (t & 7) * 8;
      const float* wp = W + (size_t)(k0 + kk) * D_H + n0 + col0;
#pragma unroll
      for (int i = 0; i < 8; ++i) ldsW[col0 + i][kk] = (_Float16)wp[i];
    }
    __syncthreads();

    v16h a;
    {
      const _Float16* p = arow + k0 + 8 * half;
#pragma unroll
      for (int i = 0; i < 8; ++i) { a[i] = p[i]; a[i + 8] = p[i + 16]; }
    }
#pragma unroll
    for (int j = 0; j < 4; ++j) {
      const _Float16* bp = &ldsW[j * 16 + ln16][8 * half];
      v16h b;
#pragma unroll
      for (int i = 0; i < 16; ++i) b[i] = bp[i < 8 ? i : i + 8];
      acc[j] = wmma_f16(a, b, acc[j]);
    }
  }

#pragma unroll
  for (int j = 0; j < 4; ++j) {
    const int cg = n0 + j * 16 + ln16;
    const float bo = bias[cg];
#pragma unroll
    for (int r = 0; r < 8; ++r) {
      const int rg = m0 + 8 * half + r;
      out[(size_t)rg * D_H + cg] = acc[j][r] + bo;
    }
  }
}

// ---------------------------------------------------------------------------
extern "C" void kernel_launch(void* const* d_in, const int* in_sizes, int n_in,
                              void* d_out, int out_size, void* d_ws, size_t ws_size,
                              hipStream_t stream) {
  (void)in_sizes; (void)n_in; (void)out_size; (void)ws_size;

  const float* v    = (const float*)d_in[0];
  const float* k    = (const float*)d_in[1];
  const float* q    = (const float*)d_in[2];
  const int*   mask = (const int*)  d_in[3];
  const float* w_v  = (const float*)d_in[4];
  const float* w_k  = (const float*)d_in[5];
  const float* w_q  = (const float*)d_in[6];
  const float* w_o  = (const float*)d_in[7];
  const float* b_o  = (const float*)d_in[8];
  float* out = (float*)d_out;

  // workspace: Q^T | Kh | V^T | attn  (each bz*head*seq*64 f16 = 12.6 MB)
  const size_t NBH = (size_t)BZ * NHEAD * SEQ * DA;
  _Float16* qt   = (_Float16*)d_ws;
  _Float16* kh   = qt + NBH;
  _Float16* vt   = kh + NBH;
  _Float16* attn = vt + NBH;

  dim3 blk(256);
  dim3 gproj(8192 / 128, D_H / 64);   // (64, 12)

  // fold 1/sqrt(d_attn) = 1/8 into Q projection
  proj_kernel<1><<<gproj, blk, 0, stream>>>(q, w_q, qt, 0.125f);
  proj_kernel<0><<<gproj, blk, 0, stream>>>(k, w_k, kh, 1.0f);
  proj_kernel<1><<<gproj, blk, 0, stream>>>(v, w_v, vt, 1.0f);

  dim3 gfa(SEQ / 128, BZ * NHEAD);    // (32, 24)
  flash_kernel<<<gfa, blk, 0, stream>>>(qt, kh, vt, mask, attn);

  oproj_kernel<<<gproj, blk, 0, stream>>>(attn, w_o, b_o, out);
}
